// ZeRe_40767829574314
// MI455X (gfx1250) — matrix operand order repaired and checked by reading, not compiled, not run on hardware
//
#include <hip/hip_runtime.h>

#define NN 16384
#define DD 128

typedef __attribute__((ext_vector_type(16))) __bf16 v16bf;
typedef __attribute__((ext_vector_type(8)))  __bf16 v8bf;
typedef __attribute__((ext_vector_type(8)))  float  v8f;

__device__ __forceinline__ unsigned short f2bf(float f) {
    // round-to-nearest-even float -> bf16 bits
    unsigned int u = __builtin_bit_cast(unsigned int, f);
    u += 0x7fffu + ((u >> 16) & 1u);
    return (unsigned short)(u >> 16);
}

__device__ __forceinline__ v16bf cat8(v8bf lo, v8bf hi) {
    return __builtin_shufflevector(lo, hi, 0,1,2,3,4,5,6,7,8,9,10,11,12,13,14,15);
}

// Assemble a 16-element bf16 fragment from two 16-byte chunks.
__device__ __forceinline__ v16bf ldfrag16(const unsigned short* p0, const unsigned short* p1) {
    v8bf lo = *(const v8bf*)p0;
    v8bf hi = *(const v8bf*)p1;
    return cat8(lo, hi);
}

// gfx1250 async DMA: 16B global -> LDS, tracked by ASYNCcnt.
// Generic pointers to LDS are {shared_aperture_hi32, lds_offset_lo32}, so the
// low 32 bits are the DS address the instruction wants.
__device__ __forceinline__ void async_b128(void* lds, const void* gmem) {
    unsigned l = (unsigned)(uintptr_t)lds;
    asm volatile("global_load_async_to_lds_b128 %0, %1, off"
                 :: "v"(l), "v"(gmem) : "memory");
}

__device__ __forceinline__ void wait_async0() {
    asm volatile("s_wait_asynccnt 0" ::: "memory");
}

// ---------------------------------------------------------------------------
// Kernel 1: projections q,k,v = items @ W + b  (f32 math, bf16 outputs).
// v is stored transposed [D, N] so attention's P@V B-fragments are contiguous.
// ---------------------------------------------------------------------------
__global__ __launch_bounds__(128) void proj_kernel(
        const float* __restrict__ items,
        const float* __restrict__ Wq, const float* __restrict__ bq,
        const float* __restrict__ Wk, const float* __restrict__ bk,
        const float* __restrict__ Wv, const float* __restrict__ bv,
        unsigned short* __restrict__ qbf,
        unsigned short* __restrict__ kbf,
        unsigned short* __restrict__ vtbf) {
    __shared__ float row[DD];
    const int i = blockIdx.x;
    const int d = threadIdx.x;
    row[d] = __builtin_nontemporal_load(&items[(size_t)i * DD + d]);
    __syncthreads();
    float aq = bq[d], ak = bk[d], av = bv[d];
    #pragma unroll 8
    for (int k = 0; k < DD; ++k) {
        float x = row[k];
        aq = fmaf(x, Wq[k * DD + d], aq);
        ak = fmaf(x, Wk[k * DD + d], ak);
        av = fmaf(x, Wv[k * DD + d], av);
    }
    qbf[(size_t)i * DD + d]  = f2bf(aq);
    kbf[(size_t)i * DD + d]  = f2bf(ak);
    vtbf[(size_t)d * NN + i] = f2bf(av);
}

// ---------------------------------------------------------------------------
// Kernel 2: flash-attention. 4 waves/block, one wave per 16 query rows.
// Per iteration (32 keys): K and V^T tiles are DMA'd once per block into
// double-buffered LDS (global_load_async_to_lds_b128 / s_wait_asynccnt),
// then: 8 WMMAs for S = Q K^T, mask*1/sqrt(D) from a non-temporal HBM
// stream, online softmax, 8 WMMAs for O += P V.
//
// Wave32 WMMA layouts (CDNA5 ISA 7.12.2):
//  A 16x32 bf16 : lane m=lane%16; elems 0..7 -> K=(lane<16?0:8)+e,
//                 elems 8..15 -> K=(lane<16?16:24)+(e-8)   (+32 per frag)
//  B 32x16 bf16 : lane n=lane%16; elems 0..7 -> K=(lane<16?0:8)+e, 8..15 -> K=(lane<16?16:24)+(e-8)
//  C/D 16x16 f32: component r -> row M=r+(lane<16?0:8), col N=lane%16
// LDS rows padded (K:136, V/P:40 halfwords) => 16 fragment lanes hit 16 banks.
// ---------------------------------------------------------------------------
__global__ __launch_bounds__(128) void attn_fa(
        const unsigned short* __restrict__ qbf,
        const unsigned short* __restrict__ kbf,
        const unsigned short* __restrict__ vtbf,
        const float* __restrict__ mask,
        float* __restrict__ out) {
    const int t     = threadIdx.x;
    const int lane  = t & 31;
    const int wv    = t >> 5;                 // wave in block (0..3)
    const int ln    = lane & 15;
    const int hl    = lane >> 4;              // half-wave select
    const int qbase = blockIdx.x * 64 + wv * 16;

    __shared__ __align__(16) unsigned short Kt[2][32][136];  // 32 keys x 128 dims
    __shared__ __align__(16) unsigned short Vt[2][DD][40];   // 128 dims x 32 keys
    __shared__ __align__(16) unsigned short Pt[4][16][40];   // per-wave P tile

    // Issue one 32-key tile (K: 8KB, V^T: 8KB) as 8x b128 per thread.
    auto issue_tile = [&](int buf, int jb) {
        #pragma unroll
        for (int i = 0; i < 4; ++i) {
            int ch = t + i * 128;             // 0..511
            int kr = ch >> 4, kc = ch & 15;   // K: 32 rows x 16 chunks
            async_b128(&Kt[buf][kr][kc * 8],
                       kbf + (size_t)(jb + kr) * DD + kc * 8);
            int vr = ch >> 2, vc = ch & 3;    // V: 128 rows x 4 chunks
            async_b128(&Vt[buf][vr][vc * 8],
                       vtbf + (size_t)vr * NN + jb + vc * 8);
        }
    };

    // Q held in registers as 4 A-fragments (K-dim 128 = 4 x 32)
    v16bf qa[4];
    {
        const unsigned short* qr = qbf + (size_t)(qbase + ln) * DD;
        #pragma unroll
        for (int f = 0; f < 4; ++f)
            qa[f] = ldfrag16(qr + f * 32 + hl * 8, qr + f * 32 + 16 + hl * 8);
    }

    v8f o[8];
    #pragma unroll
    for (int c = 0; c < 8; ++c)
        #pragma unroll
        for (int r = 0; r < 8; ++r) o[c][r] = 0.0f;

    float mrun[8], lrun[8];
    #pragma unroll
    for (int r = 0; r < 8; ++r) { mrun[r] = -3.0e38f; lrun[r] = 0.0f; }

    const float scale = 0.08838834764831845f;   // 1/sqrt(128)

    issue_tile(0, 0);
    int cur = 0;

    for (int jb = 0; jb < NN; jb += 32) {
        wait_async0();        // this wave's DMA for buf[cur] done
        __syncthreads();      // => every wave's DMA for buf[cur] done,
                              //    and everyone is finished reading buf[cur^1]
        int jn = jb + 32;
        if (jn >= NN) jn = 0; // dummy (harmless) prefetch on last iteration
        issue_tile(cur ^ 1, jn);

        // ---- S = Q K^T for key tiles [jb, jb+16) and [jb+16, jb+32) ----
        v8f s0, s1;
        #pragma unroll
        for (int r = 0; r < 8; ++r) { s0[r] = 0.0f; s1[r] = 0.0f; }

        #pragma unroll
        for (int f = 0; f < 4; ++f) {
            const unsigned short* k0 = &Kt[cur][ln][f * 32 + hl * 8];
            const unsigned short* k1 = &Kt[cur][16 + ln][f * 32 + hl * 8];
            v16bf kb0 = ldfrag16(k0, k0 + 16);
            v16bf kb1 = ldfrag16(k1, k1 + 16);
            s0 = __builtin_amdgcn_wmma_f32_16x16x32_bf16(
                     false, qa[f], false, kb0, (short)0, s0, false, false);
            s1 = __builtin_amdgcn_wmma_f32_16x16x32_bf16(
                     false, qa[f], false, kb1, (short)0, s1, false, false);
        }

        // ---- mask * scale + online softmax (mask streamed NT from HBM) ----
        const float* mrow = mask + (size_t)(qbase + hl * 8) * NN + jb + ln;
        float p0[8], p1[8], sfv[8];
        #pragma unroll
        for (int r = 0; r < 8; ++r) {
            float mk0 = __builtin_nontemporal_load(&mrow[(size_t)r * NN]);
            float mk1 = __builtin_nontemporal_load(&mrow[(size_t)r * NN + 16]);
            float v0 = s0[r] * mk0 * scale;
            float v1 = s1[r] * mk1 * scale;
            float tm = fmaxf(v0, v1);
            #pragma unroll
            for (int off = 8; off >= 1; off >>= 1)
                tm = fmaxf(tm, __shfl_xor(tm, off, 16));
            float mn = fmaxf(mrun[r], tm);
            float sf = __expf(mrun[r] - mn);
            float e0 = __expf(v0 - mn);
            float e1 = __expf(v1 - mn);
            float rs = e0 + e1;
            #pragma unroll
            for (int off = 8; off >= 1; off >>= 1)
                rs += __shfl_xor(rs, off, 16);
            mrun[r] = mn;
            lrun[r] = lrun[r] * sf + rs;
            p0[r] = e0; p1[r] = e1; sfv[r] = sf;
        }

        // rescale accumulator rows
        #pragma unroll
        for (int c = 0; c < 8; ++c)
            #pragma unroll
            for (int r = 0; r < 8; ++r) o[c][r] *= sfv[r];

        // ---- transpose P from C-layout to A-layout through LDS ----
        #pragma unroll
        for (int r = 0; r < 8; ++r) {
            Pt[wv][hl * 8 + r][ln]      = f2bf(p0[r]);
            Pt[wv][hl * 8 + r][ln + 16] = f2bf(p1[r]);
        }
        __builtin_amdgcn_wave_barrier();   // same-wave LDS ops are in-order

        v16bf pa = ldfrag16(&Pt[wv][ln][hl * 8], &Pt[wv][ln][16 + hl * 8]);

        // ---- O += P V from the LDS V^T tile ----
        #pragma unroll
        for (int c = 0; c < 8; ++c) {
            const unsigned short* vp = &Vt[cur][c * 16 + ln][hl * 8];
            v16bf vb = ldfrag16(vp, vp + 16);
            o[c] = __builtin_amdgcn_wmma_f32_16x16x32_bf16(
                       false, pa, false, vb, (short)0, o[c], false, false);
        }

        cur ^= 1;
    }

    // ---- epilogue: divide by softmax denominator, store f32 output ----
    #pragma unroll
    for (int c = 0; c < 8; ++c) {
        #pragma unroll
        for (int r = 0; r < 8; ++r) {
            float val = o[c][r] / lrun[r];
            __builtin_nontemporal_store(
                val, &out[(size_t)(qbase + hl * 8 + r) * DD + c * 16 + ln]);
        }
    }
}

extern "C" void kernel_launch(void* const* d_in, const int* in_sizes, int n_in,
                              void* d_out, int out_size, void* d_ws, size_t ws_size,
                              hipStream_t stream) {
    const float* items = (const float*)d_in[0];
    const float* mask  = (const float*)d_in[1];
    const float* Wq    = (const float*)d_in[2];
    const float* bq    = (const float*)d_in[3];
    const float* Wk    = (const float*)d_in[4];
    const float* bk    = (const float*)d_in[5];
    const float* Wv    = (const float*)d_in[6];
    const float* bv    = (const float*)d_in[7];
    float* out = (float*)d_out;

    // workspace: qbf | kbf | vt(bf16, transposed)  -> 3 * N*D * 2B = 12.6 MB
    unsigned short* qbf  = (unsigned short*)d_ws;
    unsigned short* kbf  = qbf + (size_t)NN * DD;
    unsigned short* vtbf = kbf + (size_t)NN * DD;

    proj_kernel<<<NN, DD, 0, stream>>>(items, Wq, bq, Wk, bk, Wv, bv,
                                       qbf, kbf, vtbf);
    // 256 blocks x 4 waves x 16 rows = 16384 query rows
    attn_fa<<<NN / 64, 128, 0, stream>>>(qbf, kbf, vtbf, mask, out);
}
